// CausalSelfAttention_12824772346355
// MI455X (gfx1250) — matrix operand order repaired and checked by reading, not compiled, not run on hardware
//
#include <hip/hip_runtime.h>
#include <hip/hip_bf16.h>

// ---------------------------------------------------------------------------
// CausalSelfAttention for MI455X (gfx1250): f16 WMMA flash-attention pipeline
//   T=4096, C=1024, H=16, D=64
//   - QKV + output projections: wave-level 64x128 WMMA tiles (f32 accum)
//   - attention: flash (online softmax), one wave per (head, 64-query block)
// ---------------------------------------------------------------------------

typedef __attribute__((ext_vector_type(16))) _Float16 v16h;
typedef __attribute__((ext_vector_type(8)))  _Float16 v8h;
typedef __attribute__((ext_vector_type(8)))  float    v8f;

#define T_SEQ 4096
#define C_DIM 1024
#define N_HEAD 16
#define D_HEAD 64

__device__ __forceinline__ v16h join16(v8h a, v8h b) {
  return __builtin_shufflevector(a, b, 0,1,2,3,4,5,6,7,8,9,10,11,12,13,14,15);
}

__device__ __forceinline__ v8f wmma16(v16h a, v16h b, v8f c) {
  // v_wmma_f32_16x16x32_f16: D = A(16x32 f16) * B(32x16 f16) + C(16x16 f32)
  return __builtin_amdgcn_wmma_f32_16x16x32_f16(false, a, false, b, (short)0, c,
                                                false, false);
}

// A-matrix (16x32 f16) fragment from row-major [ld] memory.
// lane l<16 : row M=l,   K = {0..7, 16..23}
// lane l>=16: row M=l-16,K = {8..15, 24..31}
__device__ __forceinline__ v16h load_a(const _Float16* __restrict__ base,
                                       int row0, int k0, int ld, int lane) {
  const int m    = row0 + (lane & 15);
  const int koff = (lane >> 4) * 8;
  const _Float16* p = base + (size_t)m * ld + k0 + koff;
  v8h lo = *(const v8h*)p;
  v8h hi = *(const v8h*)(p + 16);
  return join16(lo, hi);
}

// B-matrix (32x16 f16) fragment. B's column n is memory row (n0+n), K contiguous.
// lane l<16 : col N=l, K = {0..7, 16..23} ; lane l>=16: col N=l-16, K = {8..15, 24..31}
__device__ __forceinline__ v16h load_b(const _Float16* __restrict__ base,
                                       int n0, int k0, int ld, int lane) {
  const int n = n0 + (lane & 15);
  const int k = k0 + (lane >> 4) * 8;
  const _Float16* p = base + (size_t)n * ld + k;
  v8h lo = *(const v8h*)p;
  v8h hi = *(const v8h*)(p + 16);
  return join16(lo, hi);
}

// ---------------------------------------------------------------------------
// fp32 -> f16 conversion (grid-stride)
// ---------------------------------------------------------------------------
__global__ void cvt_f32_to_f16(const float* __restrict__ in,
                               _Float16* __restrict__ out, int n) {
  int i = blockIdx.x * blockDim.x + threadIdx.x;
  int stride = gridDim.x * blockDim.x;
  for (; i < n; i += stride) out[i] = (_Float16)in[i];
}

// ---------------------------------------------------------------------------
// Fused QKV projection: Y = x @ W.T + b  (one wave -> 64x128 output tile)
// z = blockIdx.z selects {Q,K,V}. Q,K stored (H,T,D); V stored (H,D,T).
// ---------------------------------------------------------------------------
__global__ __launch_bounds__(128)
void qkv_gemm(const _Float16* __restrict__ xh,
              const _Float16* __restrict__ wqh,
              const _Float16* __restrict__ wkh,
              const _Float16* __restrict__ wvh,
              const float* __restrict__ bq,
              const float* __restrict__ bk,
              const float* __restrict__ bv,
              _Float16* __restrict__ Qb,
              _Float16* __restrict__ Kb,
              _Float16* __restrict__ Vt) {
  const int lane  = threadIdx.x & 31;
  const int wave  = threadIdx.x >> 5;
  const int nbase = blockIdx.x * 128;
  const int mbase = (blockIdx.y * 4 + wave) * 64;
  const int z     = blockIdx.z;
  const _Float16* W    = (z == 0) ? wqh : (z == 1) ? wkh : wvh;
  const float*    bias = (z == 0) ? bq  : (z == 1) ? bk  : bv;

  v8f acc[4][8] = {};

  for (int k = 0; k < C_DIM; k += 32) {
    if (k + 32 < C_DIM) {
      // locality 3 -> WGP-scope prefetch (pull into near caches)
      __builtin_prefetch(xh + (size_t)(mbase + (lane & 15)) * C_DIM + k + 32, 0, 3);
      __builtin_prefetch(W  + (size_t)(nbase + (lane & 15)) * C_DIM + k + 32, 0, 3);
      __builtin_prefetch(W  + (size_t)(nbase + 64 + (lane & 15)) * C_DIM + k + 32, 0, 3);
    }
    v16h af[4], bf[8];
#pragma unroll
    for (int mt = 0; mt < 4; mt++) af[mt] = load_a(xh, mbase + 16 * mt, k, C_DIM, lane);
#pragma unroll
    for (int nt = 0; nt < 8; nt++) bf[nt] = load_b(W, nbase + 16 * nt, k, C_DIM, lane);
#pragma unroll
    for (int mt = 0; mt < 4; mt++)
#pragma unroll
      for (int nt = 0; nt < 8; nt++)
        acc[mt][nt] = wmma16(af[mt], bf[nt], acc[mt][nt]);
  }

  const int lnlo = lane & 15, hih = lane >> 4;
#pragma unroll
  for (int mt = 0; mt < 4; mt++)
#pragma unroll
    for (int nt = 0; nt < 8; nt++)
#pragma unroll
      for (int r = 0; r < 8; r++) {
        const int t = mbase + 16 * mt + r + 8 * hih;     // C-layout row
        const int n = nbase + 16 * nt + lnlo;            // C-layout col
        const float val = acc[mt][nt][r] + bias[n];
        const int h = n >> 6, d = n & 63;
        if (z == 2) {
          Vt[((size_t)(h * D_HEAD + d)) * T_SEQ + t] = (_Float16)val;
        } else {
          _Float16* O = z ? Kb : Qb;
          O[((size_t)h * T_SEQ + t) * D_HEAD + d] = (_Float16)val;
        }
      }
}

// ---------------------------------------------------------------------------
// Flash attention: one wave per (head, 64-row query block); online softmax.
// ---------------------------------------------------------------------------
__global__ __launch_bounds__(128)
void flash_attn(const _Float16* __restrict__ Qb,
                const _Float16* __restrict__ Kb,
                const _Float16* __restrict__ Vt,
                _Float16* __restrict__ yh) {
  __shared__ _Float16 lds[4][64][72];  // per-wave P tile, padded rows (16B align)

  const int lane = threadIdx.x & 31;
  const int wave = threadIdx.x >> 5;
  const int lnlo = lane & 15, hih = lane >> 4;
  const int h  = blockIdx.y;
  const int qb = blockIdx.x * 4 + wave;   // 0..63
  const int q0 = qb * 64;

  const _Float16* Qh = Qb + (size_t)h * T_SEQ * D_HEAD;
  const _Float16* Kh = Kb + (size_t)h * T_SEQ * D_HEAD;
  const _Float16* Vh = Vt + (size_t)h * D_HEAD * T_SEQ;

  // Q fragments: resident for whole KV loop. K-dim = D_HEAD = 2 steps of 32.
  v16h qf[4][2];
#pragma unroll
  for (int mt = 0; mt < 4; mt++)
#pragma unroll
    for (int ks = 0; ks < 2; ks++)
      qf[mt][ks] = load_a(Qh, q0 + 16 * mt, 32 * ks, D_HEAD, lane);

  v8f o[4][4] = {};
  float mrow[4][8], lrow[4][8];
#pragma unroll
  for (int mt = 0; mt < 4; mt++)
#pragma unroll
    for (int r = 0; r < 8; r++) { mrow[mt][r] = -3.0e38f; lrow[mt][r] = 0.f; }

  for (int kb = 0; kb <= qb; kb++) {
    const int k0 = kb * 64;

    // S = Q K^T  (B column s = K row s, contiguous D)
    v16h kf[4][2];
#pragma unroll
    for (int nt = 0; nt < 4; nt++)
#pragma unroll
      for (int ks = 0; ks < 2; ks++)
        kf[nt][ks] = load_b(Kh, k0 + 16 * nt, 32 * ks, D_HEAD, lane);

    v8f s[4][4] = {};
#pragma unroll
    for (int ks = 0; ks < 2; ks++)
#pragma unroll
      for (int mt = 0; mt < 4; mt++)
#pragma unroll
        for (int nt = 0; nt < 4; nt++)
          s[mt][nt] = wmma16(qf[mt][ks], kf[nt][ks], s[mt][nt]);

    const bool diag = (kb == qb);
#pragma unroll
    for (int mt = 0; mt < 4; mt++)
#pragma unroll
      for (int nt = 0; nt < 4; nt++)
#pragma unroll
        for (int r = 0; r < 8; r++) {
          float v = s[mt][nt][r] * 0.125f;  // 1/sqrt(64)
          if (diag) {
            const int rr = 16 * mt + r + 8 * hih;  // local query row
            const int cc = 16 * nt + lnlo;         // local key col
            if (cc > rr) v = -3.0e38f;
          }
          s[mt][nt][r] = v;
        }

    // Online softmax. Row of a 16x16 C tile spans 16 lanes -> xor reductions.
#pragma unroll
    for (int mt = 0; mt < 4; mt++) {
#pragma unroll
      for (int r = 0; r < 8; r++) {
        float mx = s[mt][0][r];
#pragma unroll
        for (int nt = 1; nt < 4; nt++) mx = fmaxf(mx, s[mt][nt][r]);
        mx = fmaxf(mx, __shfl_xor(mx, 1));
        mx = fmaxf(mx, __shfl_xor(mx, 2));
        mx = fmaxf(mx, __shfl_xor(mx, 4));
        mx = fmaxf(mx, __shfl_xor(mx, 8));
        const float mold = mrow[mt][r];
        const float mnew = fmaxf(mold, mx);
        const float sc   = __expf(mold - mnew);
        mrow[mt][r] = mnew;
        float rs = 0.f;
#pragma unroll
        for (int nt = 0; nt < 4; nt++) {
          const float e = __expf(s[mt][nt][r] - mnew);
          s[mt][nt][r] = e;
          rs += e;
        }
        rs += __shfl_xor(rs, 1);
        rs += __shfl_xor(rs, 2);
        rs += __shfl_xor(rs, 4);
        rs += __shfl_xor(rs, 8);
        lrow[mt][r] = lrow[mt][r] * sc + rs;
#pragma unroll
        for (int nt = 0; nt < 4; nt++) o[mt][nt][r] *= sc;
      }
    }

    // C-layout -> A-layout via per-wave LDS tile (same-wave LDS is in-order).
#pragma unroll
    for (int mt = 0; mt < 4; mt++)
#pragma unroll
      for (int nt = 0; nt < 4; nt++)
#pragma unroll
        for (int r = 0; r < 8; r++)
          lds[wave][16 * mt + r + 8 * hih][16 * nt + lnlo] =
              (_Float16)s[mt][nt][r];

    v16h pf[4][2];
#pragma unroll
    for (int mt = 0; mt < 4; mt++)
#pragma unroll
      for (int ks = 0; ks < 2; ks++) {
        const _Float16* pr = &lds[wave][16 * mt + lnlo][32 * ks + 8 * hih];
        v8h lo = *(const v8h*)pr;
        v8h hi = *(const v8h*)(pr + 16);
        pf[mt][ks] = join16(lo, hi);
      }

    // V fragments: Vt is (D,T) per head, so B column d is contiguous in key t.
    v16h vf[4][2];
#pragma unroll
    for (int nt = 0; nt < 4; nt++)
#pragma unroll
      for (int ks = 0; ks < 2; ks++)
        vf[nt][ks] = load_b(Vh, 16 * nt, k0 + 32 * ks, T_SEQ, lane);

#pragma unroll
    for (int ks = 0; ks < 2; ks++)
#pragma unroll
      for (int mt = 0; mt < 4; mt++)
#pragma unroll
        for (int nt = 0; nt < 4; nt++)
          o[mt][nt] = wmma16(pf[mt][ks], vf[nt][ks], o[mt][nt]);
  }

  // Normalize and write y (T, C) with head-interleaved columns.
#pragma unroll
  for (int mt = 0; mt < 4; mt++)
#pragma unroll
    for (int nt = 0; nt < 4; nt++)
#pragma unroll
      for (int r = 0; r < 8; r++) {
        const int t = q0 + 16 * mt + r + 8 * hih;
        const int n = h * D_HEAD + 16 * nt + lnlo;
        yh[(size_t)t * C_DIM + n] = (_Float16)(o[mt][nt][r] / lrow[mt][r]);
      }
}

// ---------------------------------------------------------------------------
// Output projection: out = y @ Wp.T + bp (fp32 output), 64x128 wave tiles
// ---------------------------------------------------------------------------
__global__ __launch_bounds__(128)
void out_gemm(const _Float16* __restrict__ yh,
              const _Float16* __restrict__ wph,
              const float* __restrict__ bp,
              float* __restrict__ out) {
  const int lane  = threadIdx.x & 31;
  const int wave  = threadIdx.x >> 5;
  const int nbase = blockIdx.x * 128;
  const int mbase = (blockIdx.y * 4 + wave) * 64;

  v8f acc[4][8] = {};

  for (int k = 0; k < C_DIM; k += 32) {
    if (k + 32 < C_DIM) {
      __builtin_prefetch(yh  + (size_t)(mbase + (lane & 15)) * C_DIM + k + 32, 0, 3);
      __builtin_prefetch(wph + (size_t)(nbase + (lane & 15)) * C_DIM + k + 32, 0, 3);
      __builtin_prefetch(wph + (size_t)(nbase + 64 + (lane & 15)) * C_DIM + k + 32, 0, 3);
    }
    v16h af[4], bf[8];
#pragma unroll
    for (int mt = 0; mt < 4; mt++) af[mt] = load_a(yh, mbase + 16 * mt, k, C_DIM, lane);
#pragma unroll
    for (int nt = 0; nt < 8; nt++) bf[nt] = load_b(wph, nbase + 16 * nt, k, C_DIM, lane);
#pragma unroll
    for (int mt = 0; mt < 4; mt++)
#pragma unroll
      for (int nt = 0; nt < 8; nt++)
        acc[mt][nt] = wmma16(af[mt], bf[nt], acc[mt][nt]);
  }

  const int lnlo = lane & 15, hih = lane >> 4;
#pragma unroll
  for (int mt = 0; mt < 4; mt++)
#pragma unroll
    for (int nt = 0; nt < 8; nt++)
#pragma unroll
      for (int r = 0; r < 8; r++) {
        const int t = mbase + 16 * mt + r + 8 * hih;
        const int n = nbase + 16 * nt + lnlo;
        out[(size_t)t * C_DIM + n] = acc[mt][nt][r] + bp[n];
      }
}

// ---------------------------------------------------------------------------
// Host launcher
// ---------------------------------------------------------------------------
extern "C" void kernel_launch(void* const* d_in, const int* in_sizes, int n_in,
                              void* d_out, int out_size, void* d_ws, size_t ws_size,
                              hipStream_t stream) {
  const float* x  = (const float*)d_in[0];
  const float* Wq = (const float*)d_in[1];
  const float* bq = (const float*)d_in[2];
  const float* Wk = (const float*)d_in[3];
  const float* bk = (const float*)d_in[4];
  const float* Wv = (const float*)d_in[5];
  const float* bv = (const float*)d_in[6];
  const float* Wp = (const float*)d_in[7];
  const float* bp = (const float*)d_in[8];

  char* ws = (char*)d_ws;
  const size_t MB = 1u << 20;
  _Float16* xh  = (_Float16*)(ws + 0 * MB);    // 8 MiB
  _Float16* wqh = (_Float16*)(ws + 8 * MB);    // 2 MiB
  _Float16* wkh = (_Float16*)(ws + 10 * MB);   // 2 MiB
  _Float16* wvh = (_Float16*)(ws + 12 * MB);   // 2 MiB
  _Float16* wph = (_Float16*)(ws + 14 * MB);   // 2 MiB
  _Float16* Qb  = (_Float16*)(ws + 16 * MB);   // 8 MiB (H,T,D)
  _Float16* Kb  = (_Float16*)(ws + 24 * MB);   // 8 MiB (H,T,D)
  _Float16* Vt  = (_Float16*)(ws + 32 * MB);   // 8 MiB (H,D,T)
  _Float16* yh  = (_Float16*)(ws + 40 * MB);   // 8 MiB (T,C)

  const int nX = T_SEQ * C_DIM;   // 4,194,304
  const int nW = C_DIM * C_DIM;   // 1,048,576
  cvt_f32_to_f16<<<2048, 256, 0, stream>>>(x,  xh,  nX);
  cvt_f32_to_f16<<<1024, 256, 0, stream>>>(Wq, wqh, nW);
  cvt_f32_to_f16<<<1024, 256, 0, stream>>>(Wk, wkh, nW);
  cvt_f32_to_f16<<<1024, 256, 0, stream>>>(Wv, wvh, nW);
  cvt_f32_to_f16<<<1024, 256, 0, stream>>>(Wp, wph, nW);

  // QKV: grid = (C/128, T/256, 3), 4 waves/block, each wave a 64x128 tile
  qkv_gemm<<<dim3(C_DIM / 128, T_SEQ / 256, 3), 128, 0, stream>>>(
      xh, wqh, wkh, wvh, bq, bk, bv, Qb, Kb, Vt);

  // Flash attention: grid = (T/256, H), wave = one 64-row query block
  flash_attn<<<dim3(T_SEQ / 256, N_HEAD), 128, 0, stream>>>(Qb, Kb, Vt, yh);

  // Output projection
  out_gemm<<<dim3(C_DIM / 128, T_SEQ / 256), 128, 0, stream>>>(yh, wph, bp,
                                                               (float*)d_out);
}
